// GraphAttentionLayer_49976239456779
// MI455X (gfx1250) — matrix operand order repaired and checked by reading, not compiled, not run on hardware
//
#include <hip/hip_runtime.h>

typedef __attribute__((ext_vector_type(16))) _Float16 v16h;
typedef __attribute__((ext_vector_type(8)))  float    v8f;

#define NN      4096      // nodes
#define INF_    256       // in features
#define OUTF    128       // out features
#define NH      4         // heads
#define NCOLS   528       // 4*128 + 4 denom + 12 zero pad  (33 tiles of 16)
#define NTILES  33
#define KTILES  128       // 4096 / 32
#define NGROUPS 3
#define TPG     11        // n-tiles per wave (3*11 = 33)
#define KSPLIT  2         // K split for occupancy; partials reduced in finish

// ---- K1: x_t = x @ w^T  (f32, exact) --------------------------------------
__global__ void gat_linear(const float* __restrict__ x, const float* __restrict__ w,
                           float* __restrict__ xt) {
  __shared__ float xs[INF_];
  const int i = blockIdx.x;
  const int d = threadIdx.x;                       // 0..127
  ((float2*)xs)[d] = ((const float2*)(x + (size_t)i * INF_))[d];
  __syncthreads();
  const float4* wr = (const float4*)(w + (size_t)d * INF_);
  float acc = 0.f;
#pragma unroll 8
  for (int k = 0; k < INF_ / 4; ++k) {
    float4 wv = wr[k];
    acc += wv.x * xs[4*k] + wv.y * xs[4*k+1] + wv.z * xs[4*k+2] + wv.w * xs[4*k+3];
  }
  xt[(size_t)i * OUTF + d] = acc;
}

// ---- K2: s[h,j] = sum_d (aw[h,j,d]+aw[h,j,d+128]) * xt[j,d] ---------------
__global__ void gat_scores(const float* __restrict__ aw, const float* __restrict__ xt,
                           float* __restrict__ s) {
  const int lane = threadIdx.x & 31;
  const int widx = blockIdx.x * (blockDim.x >> 5) + (threadIdx.x >> 5); // 0..16383
  const int h = widx >> 12, j = widx & (NN - 1);
  const float4* a = (const float4*)(aw + ((size_t)h * NN + j) * 2 * OUTF);
  const float4* xr = (const float4*)(xt + (size_t)j * OUTF);
  float4 p = a[lane], q = a[lane + 32], v = xr[lane];   // 32 lanes * 4 = 128 d
  float acc = (p.x+q.x)*v.x + (p.y+q.y)*v.y + (p.z+q.z)*v.z + (p.w+q.w)*v.w;
  for (int m = 16; m; m >>= 1) acc += __shfl_xor(acc, m, 32);
  if (lane == 0) s[(size_t)h * NN + j] = acc;
}

// ---- K2b: per-head global max ---------------------------------------------
__global__ void gat_rowmax(const float* __restrict__ s, float* __restrict__ mx) {
  __shared__ float red[256];
  const int h = blockIdx.x;
  float m = -3.0e38f;
  for (int j = threadIdx.x; j < NN; j += 256) m = fmaxf(m, s[(size_t)h * NN + j]);
  red[threadIdx.x] = m;
  __syncthreads();
  for (int w = 128; w; w >>= 1) {
    if (threadIdx.x < w) red[threadIdx.x] = fmaxf(red[threadIdx.x], red[threadIdx.x + w]);
    __syncthreads();
  }
  if (threadIdx.x == 0) mx[h] = red[0];
}

// ---- K3: build Y in f16, pre-swizzled into WMMA B-fragment layout ----------
// Storage half-index t: e = t&15 (K within 16), lane = (t>>4)&31, nt = (t>>9)%33,
// kb = t/(512*33).  j = kb*32 + (lane>=16 ? 16 : 0) + e ; col = nt*16 + (lane&15)
__global__ void gat_build_y(const float* __restrict__ s, const float* __restrict__ mx,
                            const float* __restrict__ xt, _Float16* __restrict__ y) {
  const int t = blockIdx.x * blockDim.x + threadIdx.x;   // 0 .. 128*33*512-1
  const int e = t & 15;
  const int lane = (t >> 4) & 31;
  const int nt = (t >> 9) % NTILES;
  const int kb = t / (512 * NTILES);
  const int j = kb * 32 + ((lane >= 16) ? 16 : 0) + e;
  const int c = nt * 16 + (lane & 15);
  float val = 0.f;
  if (c < NH * OUTF) {
    const int h = c >> 7, d = c & (OUTF - 1);
    val = expf(s[(size_t)h * NN + j] - mx[h]) * xt[(size_t)j * OUTF + d];
  } else if (c < NH * OUTF + NH) {
    const int h = c - NH * OUTF;
    val = expf(s[(size_t)h * NN + j] - mx[h]);
  }
  y[t] = (_Float16)val;
}

// 0/1 int -> f16 0.0/1.0 without cvt: (-v) & 0x3C00
__device__ __forceinline__ _Float16 h01(int v) {
  union { unsigned short u; _Float16 h; } cv;
  cv.u = (unsigned short)((-v) & 0x3C00);
  return cv.h;
}

// Build the 16x32 f16 A fragment from 16 raw 0/1 ints
__device__ __forceinline__ v16h cvtA(const int4& p0, const int4& p1,
                                     const int4& q0, const int4& q1) {
  v16h a;
  a[0]=h01(p0.x);  a[1]=h01(p0.y);  a[2]=h01(p0.z);  a[3]=h01(p0.w);
  a[4]=h01(p1.x);  a[5]=h01(p1.y);  a[6]=h01(p1.z);  a[7]=h01(p1.w);
  a[8]=h01(q0.x);  a[9]=h01(q0.y);  a[10]=h01(q0.z); a[11]=h01(q0.w);
  a[12]=h01(q1.x); a[13]=h01(q1.y); a[14]=h01(q1.z); a[15]=h01(q1.w);
  return a;
}

// ---- K4: OutAll_part[ks] = A[:, ksK] @ Y[ksK, :] via v_wmma_f32_16x16x32_f16
// one wave per (m-tile, n-group of 11 tiles, k-split); 256*3*2 = 1536 waves.
// Single-body K loop (#pragma unroll 1) with peeled last iteration: B loads
// clause up front (11 frags live, partial waits), unconditional next-A raw
// loads before the WMMA block, accumulators coalesce in place.
__global__ void __launch_bounds__(256)
__attribute__((amdgpu_waves_per_eu(2)))
gat_gemm(const int* __restrict__ adj, const _Float16* __restrict__ y,
         float* __restrict__ outall) {
  const int lane = threadIdx.x & 31;
  const int flat = blockIdx.x * (blockDim.x >> 5) + (threadIdx.x >> 5); // 0..1535
  const int ks = flat & (KSPLIT - 1);
  const int tile = flat >> 1;                     // 0..767
  const int mt = tile / NGROUPS;
  const int ng = tile % NGROUPS;
  const int row = mt * 16 + (lane & 15);          // A-frag: M = lane%16
  const int koff = (lane < 16) ? 0 : 16;          // lanes>=16 handle K+16
  const int kb0 = ks * (KTILES / KSPLIT);
  const int KSTEPS = KTILES / KSPLIT;

  const int* __restrict__ apos = adj + (size_t)row * NN + kb0 * 32 + koff;
  const _Float16* __restrict__ ypos =
      y + (((size_t)kb0 * NTILES + ng * TPG) * 32 + lane) * 16;

  v8f acc[TPG];
#pragma unroll
  for (int t = 0; t < TPG; ++t) acc[t] = (v8f){0.f,0.f,0.f,0.f,0.f,0.f,0.f,0.f};

  // prologue: raw A (4x b128) for the first K-step
  int4 p0 = ((const int4*)apos)[0];
  int4 p1 = ((const int4*)apos)[1];
  int4 q0 = ((const int4*)(apos + 8))[0];
  int4 q1 = ((const int4*)(apos + 8))[1];

#pragma unroll 1
  for (int kb = 0; kb < KSTEPS - 1; ++kb) {
    // B fragment batch: immediate offsets t*1024B (+16B) from one base
    v16h b[TPG];
#pragma unroll
    for (int t = 0; t < TPG; ++t)
      b[t] = *(const v16h*)(ypos + t * 512);      // 512 halves = 1024 bytes

    // convert current A while B loads are in flight
    v16h a = cvtA(p0, p1, q0, q1);

    // rotated pipeline: unconditional next-A raw loads before the WMMAs
    apos += 32;
    p0 = ((const int4*)apos)[0];
    p1 = ((const int4*)apos)[1];
    q0 = ((const int4*)(apos + 8))[0];
    q1 = ((const int4*)(apos + 8))[1];

    // warm near caches for next K-step's B region (global_prefetch_b8)
    __builtin_prefetch((const void*)(ypos + (size_t)NTILES * 512), 0, 3);

#pragma unroll
    for (int t = 0; t < TPG; ++t)
      acc[t] = __builtin_amdgcn_wmma_f32_16x16x32_f16(
          /*neg_a=*/false, a, /*neg_b=*/false, b[t],
          /*c_mod=*/(short)0, acc[t], /*reuse_a=*/false, /*reuse_b=*/false);

    ypos += (size_t)NTILES * 512;                 // next kb row of B blocks
  }

  { // peeled final K-step: no next-A reload
    v16h b[TPG];
#pragma unroll
    for (int t = 0; t < TPG; ++t)
      b[t] = *(const v16h*)(ypos + t * 512);
    v16h a = cvtA(p0, p1, q0, q1);
#pragma unroll
    for (int t = 0; t < TPG; ++t)
      acc[t] = __builtin_amdgcn_wmma_f32_16x16x32_f16(
          false, a, false, b[t], (short)0, acc[t], false, false);
  }

  // C/D layout: VGPR r, lanes 0-15 -> M=r, lanes 16-31 -> M=r+8; N = lane%16
  float* __restrict__ opart = outall + (size_t)ks * NN * NCOLS;
  const int rlo = (lane < 16) ? 0 : 8;
  const int cl = lane & 15;
#pragma unroll
  for (int t = 0; t < TPG; ++t) {
    const int nt = ng * TPG + t;
#pragma unroll
    for (int r = 0; r < 8; ++r)
      opart[(size_t)(mt * 16 + rlo + r) * NCOLS + nt * 16 + cl] = acc[t][r];
  }
}

// ---- K5: out[h,i,d] = sum_ks Num / sum_ks Den ------------------------------
__global__ void gat_finish(const float* __restrict__ outall, float* __restrict__ out) {
  const int idx = blockIdx.x * blockDim.x + threadIdx.x;  // h*NN*OUTF + i*OUTF + d
  const int d = idx & (OUTF - 1);
  const int i = (idx >> 7) & (NN - 1);
  const int h = idx >> 19;
  const float* p0 = outall;
  const float* p1 = outall + (size_t)NN * NCOLS;
  const float num = p0[(size_t)i * NCOLS + h * OUTF + d] +
                    p1[(size_t)i * NCOLS + h * OUTF + d];
  const float den = p0[(size_t)i * NCOLS + NH * OUTF + h] +
                    p1[(size_t)i * NCOLS + NH * OUTF + h];
  out[idx] = num / den;
}

extern "C" void kernel_launch(void* const* d_in, const int* in_sizes, int n_in,
                              void* d_out, int out_size, void* d_ws, size_t ws_size,
                              hipStream_t stream) {
  const float* x   = (const float*)d_in[0];  // [4096,256]
  const int*   adj = (const int*)d_in[1];    // [4096,4096] int32 0/1
  const float* w   = (const float*)d_in[2];  // [128,256]
  const float* aw  = (const float*)d_in[3];  // [4,4096,256]
  float* out = (float*)d_out;                // [4,4096,128]

  char* ws = (char*)d_ws;
  const size_t OFF_XT  = 0;                                   // 2 MB
  const size_t OFF_S   = OFF_XT + (size_t)NN * OUTF * 4;      // 64 KB
  const size_t OFF_MX  = OFF_S + (size_t)NH * NN * 4;         // 16 B (pad 256)
  const size_t OFF_Y   = OFF_MX + 256;                        // 4.125 MB
  const size_t OFF_OA  = OFF_Y + (size_t)KTILES * NTILES * 512 * 2; // 2x 8.25 MB

  float*    xt     = (float*)(ws + OFF_XT);
  float*    s      = (float*)(ws + OFF_S);
  float*    mx     = (float*)(ws + OFF_MX);
  _Float16* y      = (_Float16*)(ws + OFF_Y);
  float*    outall = (float*)(ws + OFF_OA);

  gat_linear <<<NN, 128, 0, stream>>>(x, w, xt);
  gat_scores <<<(NH * NN) / 8, 256, 0, stream>>>(aw, xt, s);
  gat_rowmax <<<NH, 256, 0, stream>>>(s, mx);
  gat_build_y<<<(KTILES * NTILES * 512) / 256, 256, 0, stream>>>(s, mx, xt, y);
  gat_gemm   <<<(256 * NGROUPS * KSPLIT) / 8, 256, 0, stream>>>(adj, y, outall);
  gat_finish <<<(NH * NN * OUTF) / 256, 256, 0, stream>>>(outall, out);
}
